// ManualLSTM_3599182594243
// MI455X (gfx1250) — compile-verified, matrix operand order checked by reading, not run on hardware
//
#include <hip/hip_runtime.h>
#include <hip/hip_bf16.h>

typedef __attribute__((ext_vector_type(16))) __bf16 v16bf;
typedef __attribute__((ext_vector_type(2)))  __bf16 v2bf;
typedef __attribute__((ext_vector_type(8)))  float  v8f;

#define B_TOT 512
#define T_LEN 1024
#define D_IN  64
#define U_DIM 128
#define BT    16      // batch tile per workgroup
#define NCOL  512     // 4 gates * U
#define NTHR  512     // 16 waves; wave w owns gate columns [32w, 32w+32)

// ---------------------------------------------------------------------------
// Prep kernel: convert W [64,128]x4 and U [128,128]x4 f32 -> bf16, transposed
// and concatenated along the gate/N axis:
//   WT[n][k] (n=0..511,k=0..63)  = W_g[k][u],  g=n>>7, u=n&127
//   UT[n][k] (n=0..511,k=0..127) = U_g[k][u]
// Transposed layout => each WMMA B-fragment lane reads 16 contiguous bf16.
// ---------------------------------------------------------------------------
__global__ void lstm_prep_weights(const float* __restrict__ Wf, const float* __restrict__ Wi,
                                  const float* __restrict__ Wc, const float* __restrict__ Wo,
                                  const float* __restrict__ Uf, const float* __restrict__ Ui,
                                  const float* __restrict__ Uc, const float* __restrict__ Uo,
                                  __bf16* __restrict__ WT, __bf16* __restrict__ UT) {
  int idx = blockIdx.x * blockDim.x + threadIdx.x;
  if (idx < NCOL * D_IN) {                       // WT part: 512*64
    int n = idx >> 6, k = idx & 63;
    int g = n >> 7, u = n & 127;
    const float* Wg = (g == 0) ? Wf : (g == 1) ? Wi : (g == 2) ? Wc : Wo;
    WT[idx] = (__bf16)Wg[k * U_DIM + u];
  } else if (idx < NCOL * D_IN + NCOL * U_DIM) { // UT part: 512*128
    int j = idx - NCOL * D_IN;
    int n = j >> 7, k = j & 127;
    int g = n >> 7, u = n & 127;
    const float* Ug = (g == 0) ? Uf : (g == 1) ? Ui : (g == 2) ? Uc : Uo;
    UT[j] = (__bf16)Ug[k * U_DIM + u];
  }
}

// ---------------------------------------------------------------------------
// Fast activations: gfx1250 has V_TANH_F32 as a hardware transcendental.
// ---------------------------------------------------------------------------
__device__ __forceinline__ float ftanh(float x) {
#if __has_builtin(__builtin_amdgcn_tanhf)
  return __builtin_amdgcn_tanhf(x);
#else
  return 1.0f - 2.0f * __fdividef(1.0f, 1.0f + __expf(2.0f * x));
#endif
}
__device__ __forceinline__ float fsigmoid(float x) {
  return 0.5f + 0.5f * ftanh(0.5f * x);
}

// ---------------------------------------------------------------------------
// Main kernel: one workgroup = 16 batch rows, full time scan, no inter-WG
// sync. Weight WMMA fragments are register-resident for the whole scan.
// ---------------------------------------------------------------------------
__global__ void __launch_bounds__(NTHR)
lstm_scan_kernel(const float* __restrict__ x,      // [B,T,D]
                 const __bf16* __restrict__ WT,    // [512,64]
                 const __bf16* __restrict__ UT,    // [512,128]
                 const float* __restrict__ bf_, const float* __restrict__ bi_,
                 const float* __restrict__ bc_, const float* __restrict__ bo_,
                 float* __restrict__ out)          // [B,U]
{
  __shared__ __align__(16) __bf16 sh[BT * U_DIM];      //  4 KB (h, bf16)
  __shared__ __align__(16) __bf16 sx[2][BT * D_IN];    //  4 KB (x_t, double buffer)
  __shared__ __align__(16) float  sg[BT * NCOL];       // 32 KB (gate pre-act)

  const int tid  = threadIdx.x;
  const int wave = tid >> 5;         // 0..15 (wave32)
  const int lane = tid & 31;
  const int b0   = blockIdx.x * BT;

  const int row = lane & 15;         // A-row / B-column within a 16-wide tile
  const int kh  = lane >> 4;         // which K-half this lane holds
  const int cb  = wave * 32;         // wave's gate-column slice base

  // --- hoist all weight B-fragments into registers (loaded once, global) ---
  v16bf wfrag[2][2];                 // [kt][nt] : x-projection, K=64
#pragma unroll
  for (int kt = 0; kt < 2; ++kt)
#pragma unroll
    for (int nt = 0; nt < 2; ++nt)
      wfrag[kt][nt] = *(const v16bf*)(WT + (cb + nt * 16 + row) * D_IN + kt * 32 + kh * 16);

  v16bf ufrag[4][2];                 // [kt][nt] : recurrent, K=128
#pragma unroll
  for (int kt = 0; kt < 4; ++kt)
#pragma unroll
    for (int nt = 0; nt < 2; ++nt)
      ufrag[kt][nt] = *(const v16bf*)(UT + (cb + nt * 16 + row) * U_DIM + kt * 32 + kh * 16);

  // --- per-lane bias for the 2 N-tiles this wave owns (constant over rows) ---
  float bv[2];
#pragma unroll
  for (int nt = 0; nt < 2; ++nt) {
    int col = cb + nt * 16 + row;
    int g = col >> 7, u = col & 127;
    const float* bg = (g == 0) ? bf_ : (g == 1) ? bi_ : (g == 2) ? bc_ : bo_;
    bv[nt] = bg[u];
  }

  // --- elementwise-phase mapping: row be, cols ub..ub+3 ; c kept in regs ---
  const int be = tid >> 5;           // 0..15
  const int ub = (tid & 31) * 4;     // 0..124
  float creg[4];
#pragma unroll
  for (int j = 0; j < 4; ++j) creg[j] = 0.0f;

  // --- x staging mapping: row bl, 2 floats at col ko ---
  const int bl = tid >> 5;
  const int ko = (tid & 31) * 2;
  const float* xrow = x + ((size_t)(b0 + bl) * T_LEN) * D_IN + ko;

  // --- init h = 0; stage x(0) into buffer 0 ---
  for (int i = tid; i < BT * U_DIM; i += NTHR) sh[i] = (__bf16)0.0f;
  {
    float2 x0 = *(const float2*)(xrow);
    v2bf p; p.x = (__bf16)x0.x; p.y = (__bf16)x0.y;
    *(v2bf*)(&sx[0][bl * D_IN + ko]) = p;
  }
  __syncthreads();

  for (int t = 0; t < T_LEN; ++t) {
    // ---- prefetch x(t+1) into registers (latency hidden under WMMA) ----
    float2 xn = make_float2(0.0f, 0.0f);
    if (t + 1 < T_LEN)
      xn = *(const float2*)(xrow + (size_t)(t + 1) * D_IN);

    // ---- matrix phase: acc = bias + x_t*W + h*U (f32 accum) ----
    v8f acc[2];
#pragma unroll
    for (int nt = 0; nt < 2; ++nt) {
      v8f a;
#pragma unroll
      for (int r = 0; r < 8; ++r) a[r] = bv[nt];
      acc[nt] = a;
    }

#pragma unroll
    for (int kt = 0; kt < 2; ++kt) {               // x part, K=64
      v16bf afrag = *(const v16bf*)(&sx[t & 1][row * D_IN + kt * 32 + kh * 16]);
#pragma unroll
      for (int nt = 0; nt < 2; ++nt)
        acc[nt] = __builtin_amdgcn_wmma_f32_16x16x32_bf16(
            false, afrag, false, wfrag[kt][nt], (short)0, acc[nt], false, false);
    }
#pragma unroll
    for (int kt = 0; kt < 4; ++kt) {               // h part, K=128
      v16bf afrag = *(const v16bf*)(&sh[row * U_DIM + kt * 32 + kh * 16]);
#pragma unroll
      for (int nt = 0; nt < 2; ++nt)
        acc[nt] = __builtin_amdgcn_wmma_f32_16x16x32_bf16(
            false, afrag, false, ufrag[kt][nt], (short)0, acc[nt], false, false);
    }

    // spill gate pre-activations: C layout row = r + 8*kh, col = cb + nt*16 + row
#pragma unroll
    for (int nt = 0; nt < 2; ++nt) {
      int col = cb + nt * 16 + row;
#pragma unroll
      for (int r = 0; r < 8; ++r)
        sg[(r + 8 * kh) * NCOL + col] = acc[nt][r];
    }
    __syncthreads();

    // ---- stage x(t+1) into the other buffer (overlaps elementwise) ----
    {
      v2bf p; p.x = (__bf16)xn.x; p.y = (__bf16)xn.y;
      *(v2bf*)(&sx[(t + 1) & 1][bl * D_IN + ko]) = p;
    }

    // ---- elementwise: c in registers (f32), h -> LDS (bf16) ----
    const float* gr = &sg[be * NCOL];
#pragma unroll
    for (int j = 0; j < 4; ++j) {
      int u = ub + j;
      float f  = fsigmoid(gr[u]);
      float ii = fsigmoid(gr[128 + u]);
      float ct = ftanh(gr[256 + u]);
      float o  = fsigmoid(gr[384 + u]);
      creg[j] = f * creg[j] + ii * ct;
      float h = o * ftanh(creg[j]);
      sh[be * U_DIM + u] = (__bf16)h;
      if (t == T_LEN - 1)
        out[(size_t)(b0 + be) * U_DIM + u] = h;
    }
    __syncthreads();   // orders sh/sx writes vs. next step's WMMA reads
  }
}

// ---------------------------------------------------------------------------
extern "C" void kernel_launch(void* const* d_in, const int* in_sizes, int n_in,
                              void* d_out, int out_size, void* d_ws, size_t ws_size,
                              hipStream_t stream) {
  (void)in_sizes; (void)n_in; (void)out_size; (void)ws_size;
  const float* x  = (const float*)d_in[0];
  const float* Wf = (const float*)d_in[1];
  const float* Uf = (const float*)d_in[2];
  const float* bf = (const float*)d_in[3];
  const float* Wi = (const float*)d_in[4];
  const float* Ui = (const float*)d_in[5];
  const float* bi = (const float*)d_in[6];
  const float* Wc = (const float*)d_in[7];
  const float* Uc = (const float*)d_in[8];
  const float* bc = (const float*)d_in[9];
  const float* Wo = (const float*)d_in[10];
  const float* Uo = (const float*)d_in[11];
  const float* bo = (const float*)d_in[12];

  __bf16* WT = (__bf16*)d_ws;                 // 512*64  bf16 =  64 KB
  __bf16* UT = WT + NCOL * D_IN;              // 512*128 bf16 = 128 KB

  const int prep_elems = NCOL * D_IN + NCOL * U_DIM;   // 98304
  lstm_prep_weights<<<(prep_elems + 255) / 256, 256, 0, stream>>>(
      Wf, Wi, Wc, Wo, Uf, Ui, Uc, Uo, WT, UT);

  lstm_scan_kernel<<<B_TOT / BT, NTHR, 0, stream>>>(
      x, WT, UT, bf, bi, bc, bo, (float*)d_out);
}
